// ResidualBlock_66262755442815
// MI455X (gfx1250) — matrix operand order repaired and checked by reading, not compiled, not run on hardware
//
#include <hip/hip_runtime.h>
#include <cmath>

typedef __attribute__((ext_vector_type(16))) _Float16 v16h;
typedef __attribute__((ext_vector_type(8)))  float    v8f;

// Problem constants
#define NB    4
#define CRES  256
#define LSEQ  2048
#define DMODEL 512
#define DINNER 1024
#define NSTATE 14
#define DTRANK 32
#define MROWS  (NB*LSEQ)          // 8192

__device__ __forceinline__ float sigmoidf_(float x) { return 1.0f / (1.0f + __expf(-x)); }
__device__ __forceinline__ float siluf_(float x)    { return x * sigmoidf_(x); }

// Convert 8 floats (two float4) scaled by s into halves d[o..o+7].
__device__ __forceinline__ void cvt8_(v16h& d, int o, float4 q0, float4 q1, float s)
{
  d[o+0] = (_Float16)(q0.x * s); d[o+1] = (_Float16)(q0.y * s);
  d[o+2] = (_Float16)(q0.z * s); d[o+3] = (_Float16)(q0.w * s);
  d[o+4] = (_Float16)(q1.x * s); d[o+5] = (_Float16)(q1.y * s);
  d[o+6] = (_Float16)(q1.z * s); d[o+7] = (_Float16)(q1.w * s);
}

// ---------------------------------------------------------------------------
// WMMA GEMM:  C[M, ldc] = A[M, lda(K)] * W[N, ldw(K)]^T  (+epilogue)
// fp32 -> f16 on load (unconditional float4 loads, 0/1-mask multiply for
// out-of-range W rows: no divergent load clauses), fp32 accumulate via
// v_wmma_f32_16x16x32_f16.  Block = 4 waves; each wave computes a 16x64
// strip (4 N-tiles) reusing one A fragment per K-chunk across 4 WMMAs.
// EPI: 0 = none, 1 = +bias, 2 = relu(+bias), 3 = softplus(+bias)
// ---------------------------------------------------------------------------
template<int EPI>
__global__ __launch_bounds__(128)
void gemm_wmma_f16(const float* __restrict__ A, const float* __restrict__ W,
                   float* __restrict__ C, const float* __restrict__ bias,
                   int Nreal, int ntiles, int K, int lda, int ldw, int ldc)
{
  const int lane = threadIdx.x & 31;
  const int wave = threadIdx.x >> 5;
  const int wt0  = (blockIdx.x * 4 + wave) * 4;   // first of 4 n-tiles
  if (wt0 >= ntiles) return;                      // wave-uniform exit
  const int mtile = blockIdx.y;
  const int hl = lane >> 4;                       // K-half select (ISA 7.12.2)
  const int mr = lane & 15;                       // row / col within tile

  const float* arow = A + (size_t)(mtile * 16 + mr) * lda;

  int          ncol[4];
  float        wmask[4];
  const float* wrow[4];
#pragma unroll
  for (int t = 0; t < 4; ++t) {
    ncol[t]  = (wt0 + t) * 16 + mr;
    const bool v = (ncol[t] < Nreal);
    wmask[t] = v ? 1.0f : 0.0f;
    wrow[t]  = W + (size_t)(v ? ncol[t] : (Nreal - 1)) * ldw;  // clamped, always valid
  }

  v8f acc[4] = {};
  for (int kc = 0; kc < K; kc += 32) {
    if (kc + 32 < K) __builtin_prefetch(arow + kc + 32 + hl * 8, 0, 0);
    const int klo = kc + hl * 8;                  // A 16x32: K 0-7/16-23 vs 8-15/24-31
    v16h a;
    cvt8_(a, 0, *(const float4*)(arow + klo),      *(const float4*)(arow + klo + 4),  1.0f);
    cvt8_(a, 8, *(const float4*)(arow + klo + 16), *(const float4*)(arow + klo + 20), 1.0f);
    const int kb = kc + hl * 8;                   // B 32x16: K 0-7/16-23 vs 8-15/24-31
#pragma unroll
    for (int t = 0; t < 4; ++t) {
      v16h b;
      cvt8_(b, 0, *(const float4*)(wrow[t] + kb),     *(const float4*)(wrow[t] + kb + 4),  wmask[t]);
      cvt8_(b, 8, *(const float4*)(wrow[t] + kb + 16), *(const float4*)(wrow[t] + kb + 20), wmask[t]);
      acc[t] = __builtin_amdgcn_wmma_f32_16x16x32_f16(false, a, false, b,
                                                      (short)0, acc[t], false, false);
    }
  }

#pragma unroll
  for (int t = 0; t < 4; ++t) {
    float bv = 0.0f;
    if (EPI != 0) {
      const int ci = ncol[t] < Nreal ? ncol[t] : (Nreal - 1);
      bv = bias[ci] * wmask[t];
    }
#pragma unroll
    for (int r = 0; r < 8; ++r) {
      const int m = mtile * 16 + hl * 8 + r;
      float v = acc[t][r];
      if (EPI == 1)      { v += bv; }
      else if (EPI == 2) { v += bv; v = v > 0.0f ? v : 0.0f; }
      else if (EPI == 3) { v += bv; v = (v > 20.0f) ? v : log1pf(__expf(v)); }
      C[(size_t)m * ldc + ncol[t]] = v;
    }
  }
}

// ---------------------------------------------------------------------------
// Dilated conv (k=3, dilation=2, 'same') as 3 row-shifted WMMA GEMMs.
// Axd = (x + d) in (b,l)-major [8192,256]; Wt = [3][512][256] (tap, out, in).
// Out-of-range taps: clamped pointer + 0/1 mask multiply (no divergent loads).
// ---------------------------------------------------------------------------
__global__ __launch_bounds__(128)
void dconv_gemm_wmma(const float* __restrict__ Axd, const float* __restrict__ Wt,
                     const float* __restrict__ dc_b, float* __restrict__ C)
{
  const int lane = threadIdx.x & 31;
  const int wave = threadIdx.x >> 5;
  const int wt0  = (blockIdx.x * 4 + wave) * 4;   // 4 of 32 n-tiles (N=512)
  const int mtile = blockIdx.y;                   // 0..511
  const int hl = lane >> 4;
  const int mr = lane & 15;
  const int row = mtile * 16 + mr;
  const int l   = row & (LSEQ - 1);

  int ncol[4];
  const float* wrow0[4];
#pragma unroll
  for (int t = 0; t < 4; ++t) {
    ncol[t]  = (wt0 + t) * 16 + mr;
    wrow0[t] = Wt + (size_t)ncol[t] * CRES;
  }

  v8f acc[4] = {};
#pragma unroll
  for (int k = 0; k < 3; ++k) {
    const int  lp    = l + 2 * k - 2;
    const bool aval  = (lp >= 0) && (lp < LSEQ);  // zero padding, in-batch
    const float amask = aval ? 1.0f : 0.0f;
    const float* arow = Axd + (size_t)(aval ? (row + 2 * k - 2) : row) * CRES;
    const float* wtap = (size_t)k * (512 * CRES) + (const float*)nullptr == nullptr ? nullptr : nullptr; (void)wtap;
    for (int kc = 0; kc < CRES; kc += 32) {
      const int klo = kc + hl * 8;
      v16h a;
      cvt8_(a, 0, *(const float4*)(arow + klo),      *(const float4*)(arow + klo + 4),  amask);
      cvt8_(a, 8, *(const float4*)(arow + klo + 16), *(const float4*)(arow + klo + 20), amask);
      const int kb = kc + hl * 8;
#pragma unroll
      for (int t = 0; t < 4; ++t) {
        const float* wr = wrow0[t] + (size_t)k * (512 * CRES);
        v16h b;
        cvt8_(b, 0, *(const float4*)(wr + kb),     *(const float4*)(wr + kb + 4),  1.0f);
        cvt8_(b, 8, *(const float4*)(wr + kb + 16), *(const float4*)(wr + kb + 20), 1.0f);
        acc[t] = __builtin_amdgcn_wmma_f32_16x16x32_f16(false, a, false, b,
                                                        (short)0, acc[t], false, false);
      }
    }
  }
#pragma unroll
  for (int t = 0; t < 4; ++t) {
    const float bv = dc_b[ncol[t]];
#pragma unroll
    for (int r = 0; r < 8; ++r) {
      const int m = mtile * 16 + hl * 8 + r;
      C[(size_t)m * 512 + ncol[t]] = acc[t][r] + bv;
    }
  }
}

// --------------------------- small / elementwise ---------------------------

__global__ void diffproj_k(const float* __restrict__ ds, const float* __restrict__ w,
                           const float* __restrict__ b, float* __restrict__ dv)
{
  const int t = blockIdx.x * 256 + threadIdx.x;   // < 1024
  const int bb = t >> 8, c = t & 255;
  float acc = b[c];
  for (int k = 0; k < 512; ++k) acc += ds[bb * 512 + k] * w[c * 512 + k];
  dv[t] = acc;                                     // dv[b*256+c]
}

__global__ void transpose_in_k(const float* __restrict__ x, const float* __restrict__ au,
                               const float* __restrict__ dv,
                               float* __restrict__ xdt, float* __restrict__ aut)
{
  const size_t idx = (size_t)blockIdx.x * 256 + threadIdx.x;  // < 4*256*2048
  const int l = (int)(idx & (LSEQ - 1));
  const int c = (int)((idx >> 11) & 255);
  const int b = (int)(idx >> 19);
  const size_t m = (size_t)b * LSEQ + l;
  xdt[m * CRES + c] = x[idx] + dv[b * CRES + c];
  aut[m * CRES + c] = au[idx];
}

__global__ void wt_transpose_k(const float* __restrict__ dcw, float* __restrict__ wt)
{
  const int idx = blockIdx.x * 256 + threadIdx.x;  // < 3*512*256
  const int k   = idx / (512 * 256);
  const int rem = idx - k * (512 * 256);
  const int o   = rem >> 8;
  const int c   = rem & 255;
  wt[idx] = dcw[(o * 256 + c) * 3 + k];            // [k][o][c] <- [o][c][k]
}

__global__ __launch_bounds__(256)
void bn_stats_k(const float* __restrict__ Y, const float* __restrict__ g,
                const float* __restrict__ be, float* __restrict__ scale,
                float* __restrict__ shift)
{
  __shared__ float s1[256], s2[256];
  const int o = blockIdx.x;                        // channel 0..511
  float a = 0.0f, q = 0.0f;
  for (int m = threadIdx.x; m < MROWS; m += 256) {
    const float v = Y[(size_t)m * 512 + o];
    a += v; q += v * v;
  }
  s1[threadIdx.x] = a; s2[threadIdx.x] = q;
  __syncthreads();
  for (int s = 128; s > 0; s >>= 1) {
    if (threadIdx.x < s) { s1[threadIdx.x] += s1[threadIdx.x + s];
                           s2[threadIdx.x] += s2[threadIdx.x + s]; }
    __syncthreads();
  }
  if (threadIdx.x == 0) {
    const float mu  = s1[0] * (1.0f / MROWS);
    const float var = s2[0] * (1.0f / MROWS) - mu * mu;
    const float rs  = rsqrtf(var + 1e-5f);
    const float sc  = g[o] * rs;
    scale[o] = sc;
    shift[o] = be[o] - mu * sc;
  }
}

__global__ void bn_apply_k(const float* __restrict__ Y, const float* __restrict__ scale,
                           const float* __restrict__ shift, float* __restrict__ out)
{
  const size_t idx = (size_t)blockIdx.x * 256 + threadIdx.x;  // < 8192*512
  const int o = (int)(idx & 511);
  out[idx] = Y[idx] * scale[o] + shift[o];
}

__global__ void dwconv_silu_k(const float* __restrict__ xz, const float* __restrict__ w,
                              const float* __restrict__ b, float* __restrict__ u)
{
  const size_t idx = (size_t)blockIdx.x * 256 + threadIdx.x;  // < 8192*1024
  const int d = (int)(idx & (DINNER - 1));
  const int m = (int)(idx >> 10);
  const int l = m & (LSEQ - 1);
  float acc = b[d];
#pragma unroll
  for (int j = 0; j < 4; ++j) {
    const int lj = l - 3 + j;                      // causal pad (3,0)
    const int lc = lj >= 0 ? lj : 0;               // clamp + mask (no divergent load)
    const float xv = xz[(size_t)(m - l + lc) * 2048 + d];
    acc += w[d * 4 + j] * xv * (lj >= 0 ? 1.0f : 0.0f);
  }
  u[idx] = siluf_(acc);
}

// ---------------------- selective scan (fwd / bwd) -------------------------
// 4096 independent (b,d) channels, one lane each; 14 fp32 states in VGPRs;
// v_exp_f32 on the trans pipe overlaps the FMA chain.
__global__ __launch_bounds__(256)
void scan_fwd_k(const float* __restrict__ u, const float* __restrict__ dt,
                const float* __restrict__ xdbl, const float* __restrict__ A_log,
                float* __restrict__ acc)
{
  const int idx = blockIdx.x * 256 + threadIdx.x;  // < 4096
  const int d = idx & (DINNER - 1);
  const int b = idx >> 10;
  float a[NSTATE], h[NSTATE];
#pragma unroll
  for (int n = 0; n < NSTATE; ++n) { a[n] = -__expf(A_log[d * NSTATE + n]); h[n] = 0.0f; }
  const size_t mb = (size_t)b * LSEQ;
  for (int l = 0; l < LSEQ; ++l) {
    const size_t m = mb + l;
    const float dtv = dt[m * DINNER + d];
    const float du  = dtv * u[m * DINNER + d];
    const float* bc = xdbl + m * 64;               // cols 32..45 = B, 46..59 = C
    float ys = 0.0f;
#pragma unroll
    for (int n = 0; n < NSTATE; ++n) {
      h[n] = __expf(dtv * a[n]) * h[n] + du * bc[32 + n];
      ys  += h[n] * bc[46 + n];
    }
    acc[m * DINNER + d] = ys;
  }
}

__global__ __launch_bounds__(256)
void scan_bwd_k(const float* __restrict__ u, const float* __restrict__ dt,
                const float* __restrict__ xdbl, const float* __restrict__ Ab_log,
                const float* __restrict__ Dp, const float* __restrict__ xz,
                float* __restrict__ accY)   // reads fwd result, writes gated sum
{
  const int idx = blockIdx.x * 256 + threadIdx.x;
  const int d = idx & (DINNER - 1);
  const int b = idx >> 10;
  float a[NSTATE], h[NSTATE];
#pragma unroll
  for (int n = 0; n < NSTATE; ++n) { a[n] = -__expf(Ab_log[d * NSTATE + n]); h[n] = 0.0f; }
  const float dd = Dp[d];
  const size_t mb = (size_t)b * LSEQ;
  for (int l = LSEQ - 1; l >= 0; --l) {
    const size_t m = mb + l;
    const float dtv = dt[m * DINNER + d];
    const float uv  = u[m * DINNER + d];
    const float du  = dtv * uv;
    const float* bc = xdbl + m * 64;
    float ys = 0.0f;
#pragma unroll
    for (int n = 0; n < NSTATE; ++n) {
      h[n] = __expf(dtv * a[n]) * h[n] + du * bc[32 + n];
      ys  += h[n] * bc[46 + n];
    }
    const float tot = accY[m * DINNER + d] + ys + 2.0f * uv * dd;  // u*D per direction
    const float zv  = xz[m * 2048 + DINNER + d];                   // z half
    accY[m * DINNER + d] = tot * siluf_(zv);
  }
}

__global__ void gate_k(const float* __restrict__ ym, const float* __restrict__ cond,
                       float* __restrict__ g)
{
  const size_t idx = (size_t)blockIdx.x * 256 + threadIdx.x;  // < 8192*256
  const int c = (int)(idx & 255);
  const size_t m = idx >> 8;
  const float gv = ym[m * 512 + c]       + cond[m * 512 + c];
  const float fv = ym[m * 512 + 256 + c] + cond[m * 512 + 256 + c];
  g[idx] = sigmoidf_(gv) * tanhf(fv);
}

__global__ void final_k(const float* __restrict__ x, const float* __restrict__ O,
                        float* __restrict__ out)
{
  const size_t idx = (size_t)blockIdx.x * 256 + threadIdx.x;  // < 4*256*2048
  const int l = (int)(idx & (LSEQ - 1));
  const int c = (int)((idx >> 11) & 255);
  const int b = (int)(idx >> 19);
  const size_t m = (size_t)b * LSEQ + l;
  out[idx] = (x[idx] + O[m * 512 + c]) * 0.70710678118654752f;   // (x+res)/sqrt(2)
  out[(size_t)NB * CRES * LSEQ + idx] = O[m * 512 + 256 + c];    // skip
}

// ---------------------------------------------------------------------------
extern "C" void kernel_launch(void* const* d_in, const int* in_sizes, int n_in,
                              void* d_out, int out_size, void* d_ws, size_t ws_size,
                              hipStream_t stream)
{
  (void)in_sizes; (void)n_in; (void)out_size; (void)ws_size;
  const float* x       = (const float*)d_in[0];
  const float* audio   = (const float*)d_in[1];
  const float* dstep   = (const float*)d_in[2];
  const float* dc_w    = (const float*)d_in[3];
  const float* dc_b    = (const float*)d_in[4];
  const float* bn_g    = (const float*)d_in[5];
  const float* bn_b    = (const float*)d_in[6];
  const float* diff_w  = (const float*)d_in[7];
  const float* diff_b  = (const float*)d_in[8];
  const float* cond_w  = (const float*)d_in[9];   // [512,256,1] -> [512,256]
  const float* cond_b  = (const float*)d_in[10];
  const float* outp_w  = (const float*)d_in[11];  // [512,256,1] -> [512,256]
  const float* outp_b  = (const float*)d_in[12];
  const float* inpj_w  = (const float*)d_in[13];  // [2048,512]
  const float* mconv_w = (const float*)d_in[14];  // [1024,1,4]
  const float* mconv_b = (const float*)d_in[15];
  const float* xproj_w = (const float*)d_in[16];  // [60,1024]
  const float* dtpj_w  = (const float*)d_in[17];  // [1024,32]
  const float* dtpj_b  = (const float*)d_in[18];
  const float* A_log   = (const float*)d_in[19];
  const float* Ab_log  = (const float*)d_in[20];
  const float* Dp      = (const float*)d_in[21];
  const float* mout_w  = (const float*)d_in[22];  // [512,1024]
  float* out = (float*)d_out;

  // Workspace layout (floats), lifetime-based reuse. Peak ~222 MB.
  float* ws = (float*)d_ws;
  constexpr size_t S_XDT = 0;                          // [8192,256]  x + d, (b,l)-major
  constexpr size_t S_AUT = S_XDT + (size_t)MROWS*CRES; // [8192,256]  audio transposed
  constexpr size_t S_DV  = S_AUT + (size_t)MROWS*CRES; // [4,256]
  constexpr size_t S_WT  = S_DV  + 1024;               // [3,512,256]
  constexpr size_t S_YC  = S_WT  + 3*512*256;          // [8192,512]  conv out (later cond_t)
  constexpr size_t S_BN  = S_YC  + (size_t)MROWS*512;  // [2,512] scale/shift
  constexpr size_t S_YBN = S_BN  + 1024;               // [8192,512]  BN out (later ym_out)
  constexpr size_t S_XZ  = S_YBN + (size_t)MROWS*512;  // [8192,2048] xz (later O)
  constexpr size_t S_U   = S_XZ  + (size_t)MROWS*2048; // [8192,1024] u
  constexpr size_t S_XD  = S_U   + (size_t)MROWS*DINNER; // [8192,64] x_dbl (padded)
  constexpr size_t S_DT  = S_XD  + (size_t)MROWS*64;   // [8192,1024] dt
  constexpr size_t S_AC  = S_DT  + (size_t)MROWS*DINNER; // [8192,1024] scan acc -> Ym

  float* xdt   = ws + S_XDT;
  float* aut   = ws + S_AUT;
  float* dv    = ws + S_DV;
  float* wt    = ws + S_WT;
  float* yc    = ws + S_YC;
  float* bns   = ws + S_BN;
  float* ybn   = ws + S_YBN;
  float* xz    = ws + S_XZ;
  float* u     = ws + S_U;
  float* xdbl  = ws + S_XD;
  float* dtb   = ws + S_DT;
  float* accY  = ws + S_AC;
  float* ymo   = ybn;          // reuse after in_proj consumed ybn
  float* condt = yc;           // reuse after BN consumed yc
  float* gbuf  = xdt;          // reuse after conv consumed xdt
  float* obuf  = xz;           // reuse (first 8192*512 floats) after scans consumed z

  // 1. diffusion-step projection, input transposes, conv-weight transpose
  diffproj_k<<<4, 256, 0, stream>>>(dstep, diff_w, diff_b, dv);
  transpose_in_k<<<(NB*CRES*LSEQ)/256, 256, 0, stream>>>(x, audio, dv, xdt, aut);
  wt_transpose_k<<<(3*512*256)/256, 256, 0, stream>>>(dc_w, wt);

  // 2. dilated conv as 3 shifted WMMA GEMMs -> [8192,512] (32 n-tiles)
  dconv_gemm_wmma<<<dim3(2, 512), 128, 0, stream>>>(xdt, wt, dc_b, yc);

  // 3. BatchNorm (batch stats) + affine
  bn_stats_k<<<512, 256, 0, stream>>>(yc, bn_g, bn_b, bns, bns + 512);
  bn_apply_k<<<(MROWS*512)/256, 256, 0, stream>>>(yc, bns, bns + 512, ybn);

  // 4. in_proj: [8192,512] x [512->2048] -> xz  (128 n-tiles)
  gemm_wmma_f16<0><<<dim3(8, 512), 128, 0, stream>>>(ybn, inpj_w, xz, nullptr,
                                                     2048, 128, 512, 512, 512, 2048);
  // 5. depthwise causal conv (k=4) + SiLU -> u
  dwconv_silu_k<<<(MROWS*DINNER)/256, 256, 0, stream>>>(xz, mconv_w, mconv_b, u);

  // 6. x_proj: [8192,1024] x [1024->60] -> x_dbl padded to 64 (4 n-tiles)
  gemm_wmma_f16<0><<<dim3(1, 512), 128, 0, stream>>>(u, xproj_w, xdbl, nullptr,
                                                     60, 4, 1024, 1024, 1024, 64);
  // 7. dt_proj + softplus: [8192,32] x [32->1024]  (64 n-tiles)
  gemm_wmma_f16<3><<<dim3(4, 512), 128, 0, stream>>>(xdbl, dtpj_w, dtb, dtpj_b,
                                                     1024, 64, 32, 64, 32, 1024);
  // 8. bidirectional selective scan, fused u*D + silu(z) gate
  scan_fwd_k<<<16, 256, 0, stream>>>(u, dtb, xdbl, A_log, accY);
  scan_bwd_k<<<16, 256, 0, stream>>>(u, dtb, xdbl, Ab_log, Dp, xz, accY);

  // 9. out_proj: [8192,1024] x [1024->512]  (32 n-tiles)
  gemm_wmma_f16<0><<<dim3(2, 512), 128, 0, stream>>>(accY, mout_w, ymo, nullptr,
                                                     512, 32, 1024, 1024, 1024, 512);
  // 10. conditioning 1x1 conv + ReLU: [8192,256] x [256->512] (32 n-tiles)
  gemm_wmma_f16<2><<<dim3(2, 512), 128, 0, stream>>>(aut, cond_w, condt, cond_b,
                                                     512, 32, 256, 256, 256, 512);
  // 11. sigmoid/tanh gating
  gate_k<<<(MROWS*CRES)/256, 256, 0, stream>>>(ymo, condt, gbuf);

  // 12. output 1x1 conv: [8192,256] x [256->512] (32 n-tiles)
  gemm_wmma_f16<1><<<dim3(2, 512), 128, 0, stream>>>(gbuf, outp_w, obuf, outp_b,
                                                     512, 32, 256, 256, 256, 512);
  // 13. residual/skip split + layout back to [B,C,L]
  final_k<<<(NB*CRES*LSEQ)/256, 256, 0, stream>>>(x, obuf, out);
}
